// CrossScanMambaBlock_45896020525188
// MI455X (gfx1250) — matrix operand order repaired and checked by reading, not compiled, not run on hardware
//
#include <hip/hip_runtime.h>

typedef __attribute__((ext_vector_type(16))) _Float16 v16h;
typedef __attribute__((ext_vector_type(8)))  _Float16 v8h;
typedef __attribute__((ext_vector_type(8)))  float    v8f;

#define BATCH   2
#define HH      56
#define WW      56
#define LSEQ    (HH*WW)      // 3136 = 98*32
#define DMODEL  256
#define DINNER  512
#define DSTATE  16
#define DTRANK  16
#define DCONV   4
#define NSCAN   4
#define NSLICE  (NSCAN*BATCH) // 8
#define XDBLW   (DTRANK + 2*DSTATE) // 48

// ---------------- f32 -> f16 convert ----------------
__global__ void k_f2h(const float* __restrict__ src, _Float16* __restrict__ dst, int n) {
  int i = blockIdx.x * blockDim.x + threadIdx.x;
  if (i < n) dst[i] = (_Float16)src[i];
}

// ---------------- RMSNorm + build 4 scan orderings (f16) ----------------
__global__ void k_rmsnorm_scan(const float* __restrict__ x, const float* __restrict__ norm_w,
                               _Float16* __restrict__ scans) {
  int bl = blockIdx.x;               // b*LSEQ + l
  int b  = bl / LSEQ;
  int l  = bl % LSEQ;
  int c  = threadIdx.x;
  float xv = x[(size_t)bl * DMODEL + c];
  float ss = xv * xv;
  #pragma unroll
  for (int off = 16; off > 0; off >>= 1) ss += __shfl_xor(ss, off, 32);
  __shared__ float red[8];
  int wid = c >> 5, lane = c & 31;
  if (lane == 0) red[wid] = ss;
  __syncthreads();
  float tot = 0.f;
  #pragma unroll
  for (int i = 0; i < 8; i++) tot += red[i];
  float nv = xv * rsqrtf(tot * (1.0f / DMODEL) + 1e-6f) * norm_w[c];
  _Float16 hv = (_Float16)nv;
  size_t sstr = (size_t)LSEQ * DMODEL;
  int hh = l / WW, ww = l % WW;
  int lt = ww * HH + hh;             // transposed position holding n[l]
  scans[(size_t)(0 * BATCH + b) * sstr + (size_t)l              * DMODEL + c] = hv;
  scans[(size_t)(1 * BATCH + b) * sstr + (size_t)(LSEQ - 1 - l) * DMODEL + c] = hv;
  scans[(size_t)(2 * BATCH + b) * sstr + (size_t)lt             * DMODEL + c] = hv;
  scans[(size_t)(3 * BATCH + b) * sstr + (size_t)(LSEQ - 1 - lt)* DMODEL + c] = hv;
}

// ------- WMMA GEMM: C[slice][M,NN] = A[slice][M,KK] * W[scan][NN,KK]^T --------
// One wave computes a (16*MT) x (16*NT) strip. KK, NN compile-time so that row
// offsets fold into the 24-bit instruction offset (2 base pointers, no spills).
// grid: (M/(16*MT), NN/(16*NT), NSLICE), block: 32 (one wave)
template <int MT, int NT, int NN, int KK>
__global__ void k_gemm_wmma(const _Float16* __restrict__ A, const _Float16* __restrict__ Wt,
                            float* __restrict__ C, int M) {
  int slice = blockIdx.z;
  int s = slice / BATCH;             // weight index (per scan, shared across batch)
  const _Float16* As = A  + (size_t)slice * M * KK;
  const _Float16* Ws = Wt + (size_t)s     * NN * KK;
  float*          Cs = C  + (size_t)slice * M * NN;
  int m0 = blockIdx.x * (16 * MT);
  int n0 = blockIdx.y * (16 * NT);
  int lane = threadIdx.x & 31;
  int hi = lane >> 4;                // half-wave select
  int lm = lane & 15;
  // A fragment (16x32 f16): lane<16 -> M=lm, chunks K=[0..7],[16..23]; lane>=16 -> K=[8..15],[24..31]
  const _Float16* aptr = As + (size_t)(m0 + lm) * KK + hi * 8;
  // B fragment (32x16 f16): N=lm, lane<16 -> chunks K=[0..7],[16..23]; lane>=16 -> K=[8..15],[24..31]
  const _Float16* wptr = Ws + (size_t)(n0 + lm) * KK + hi * 8;

  v8f acc[MT][NT];
  #pragma unroll
  for (int mt = 0; mt < MT; mt++)
    #pragma unroll
    for (int t = 0; t < NT; t++) acc[mt][t] = (v8f){};

  for (int k0 = 0; k0 < KK; k0 += 32) {
    v8h a0[MT], a1[MT], b0[NT], b1[NT];
    #pragma unroll
    for (int mt = 0; mt < MT; mt++) {
      a0[mt] = *reinterpret_cast<const v8h*>(aptr + mt * 16 * KK + k0);
      a1[mt] = *reinterpret_cast<const v8h*>(aptr + mt * 16 * KK + k0 + 16);
    }
    #pragma unroll
    for (int t = 0; t < NT; t++) {
      b0[t] = *reinterpret_cast<const v8h*>(wptr + t * 16 * KK + k0);
      b1[t] = *reinterpret_cast<const v8h*>(wptr + t * 16 * KK + k0 + 16);
    }
    if (k0 + 32 < KK) {
      #pragma unroll
      for (int mt = 0; mt < MT; mt++) __builtin_prefetch(aptr + mt * 16 * KK + k0 + 32, 0, 0);
      #pragma unroll
      for (int t = 0; t < NT; t++) __builtin_prefetch(wptr + t * 16 * KK + k0 + 32, 0, 0);
    }
    #pragma unroll
    for (int mt = 0; mt < MT; mt++) {
      v16h af = __builtin_shufflevector(a0[mt], a1[mt], 0,1,2,3,4,5,6,7,8,9,10,11,12,13,14,15);
      #pragma unroll
      for (int t = 0; t < NT; t++) {
        v16h bf = __builtin_shufflevector(b0[t], b1[t], 0,1,2,3,4,5,6,7,8,9,10,11,12,13,14,15);
        acc[mt][t] = __builtin_amdgcn_wmma_f32_16x16x32_f16(false, af, false, bf,
                                                            (short)0, acc[mt][t], false, false);
      }
    }
  }
  // C/D layout: lanes 0-15: N=lane, M=r ; lanes 16-31: N=lane-16, M=8+r
  #pragma unroll
  for (int mt = 0; mt < MT; mt++) {
    int row0 = m0 + mt * 16 + hi * 8;
    #pragma unroll
    for (int t = 0; t < NT; t++) {
      int col = n0 + t * 16 + lm;
      #pragma unroll
      for (int r = 0; r < 8; r++) Cs[(size_t)(row0 + r) * NN + col] = acc[mt][t][r];
    }
  }
}

// ---------------- depthwise causal conv1d (k=4) + SiLU ----------------
__global__ void k_conv_silu(const float* __restrict__ xz, const float* __restrict__ convW,
                            const float* __restrict__ convb, float* __restrict__ u,
                            _Float16* __restrict__ uh) {
  size_t idx = (size_t)blockIdx.x * blockDim.x + threadIdx.x;
  size_t total = (size_t)NSLICE * LSEQ * DINNER;
  if (idx >= total) return;
  int d = (int)(idx % DINNER);
  size_t rem = idx / DINNER;
  int l = (int)(rem % LSEQ);
  int slice = (int)(rem / LSEQ);
  int s = slice / BATCH;
  const float* xzs = xz + (size_t)slice * LSEQ * (2 * DINNER);
  const float* wv  = convW + ((size_t)s * DINNER + d) * DCONV;
  float acc = convb[s * DINNER + d];
  #pragma unroll
  for (int j = 0; j < DCONV; j++) {
    int lp = l + j - (DCONV - 1);
    if (lp >= 0) acc += xzs[(size_t)lp * (2 * DINNER) + d] * wv[j];
  }
  float uv = acc / (1.0f + __expf(-acc));   // silu
  u[idx]  = uv;
  uh[idx] = (_Float16)uv;
}

// ---------------- selective scan (sequential over L), fused dt+softplus+gate ----------
__global__ void k_scan(const float* __restrict__ xdbl, const float* __restrict__ u,
                       const float* __restrict__ xz,
                       const float* __restrict__ Wdt, const float* __restrict__ bdt,
                       const float* __restrict__ A_log, const float* __restrict__ Dp,
                       _Float16* __restrict__ y) {
  int slice = blockIdx.x;
  int s = slice / BATCH;
  int d = threadIdx.x;
  size_t sd = (size_t)s * DINNER + d;
  float wdt[DTRANK], Ar[DSTATE], h[DSTATE];
  #pragma unroll
  for (int r = 0; r < DTRANK; r++) wdt[r] = Wdt[sd * DTRANK + r];
  float bdtv = bdt[sd];
  #pragma unroll
  for (int n = 0; n < DSTATE; n++) { Ar[n] = -__expf(A_log[sd * DSTATE + n]); h[n] = 0.f; }
  float Dpv = Dp[sd];
  const float* xds = xdbl + (size_t)slice * LSEQ * XDBLW;
  const float* us  = u    + (size_t)slice * LSEQ * DINNER;
  const float* zs  = xz   + (size_t)slice * LSEQ * (2 * DINNER) + DINNER;
  _Float16*    ys  = y    + (size_t)slice * LSEQ * DINNER;
  __shared__ float sh[2][XDBLW];     // double buffer: one sync per step
  if (d < XDBLW) sh[0][d] = xds[d];
  __syncthreads();
  for (int t = 0; t < LSEQ; t++) {
    int cur = t & 1;
    if (d < XDBLW && (t + 1) < LSEQ) sh[cur ^ 1][d] = xds[(size_t)(t + 1) * XDBLW + d];
    const float* row = sh[cur];
    float dtv = bdtv;
    #pragma unroll
    for (int r = 0; r < DTRANK; r++) dtv += row[r] * wdt[r];
    dtv = (dtv > 20.f) ? dtv : log1pf(__expf(dtv));   // softplus
    float ut = us[(size_t)t * DINNER + d];
    float du = dtv * ut;
    float yv = 0.f;
    #pragma unroll
    for (int n = 0; n < DSTATE; n++) {
      h[n] = h[n] * __expf(dtv * Ar[n]) + du * row[DTRANK + n];       // B_t
      yv  += h[n] * row[DTRANK + DSTATE + n];                          // C_t
    }
    float zv = zs[(size_t)t * (2 * DINNER) + d];
    float yo = (yv + ut * Dpv) * (zv / (1.0f + __expf(-zv)));
    ys[(size_t)t * DINNER + d] = (_Float16)yo;
    __syncthreads();
  }
}

// ---------------- weighted combine of the 4 scan outputs + residual ----------------
__global__ void k_combine(const float* __restrict__ x, const float* __restrict__ outs,
                          const float* __restrict__ logits, float* __restrict__ out) {
  size_t idx = (size_t)blockIdx.x * blockDim.x + threadIdx.x;
  size_t total = (size_t)BATCH * LSEQ * DMODEL;
  if (idx >= total) return;
  int c = (int)(idx % DMODEL);
  size_t rem = idx / DMODEL;
  int l = (int)(rem % LSEQ);
  int b = (int)(rem / LSEQ);
  float l0 = logits[0], l1 = logits[1], l2 = logits[2], l3 = logits[3];
  float mx = fmaxf(fmaxf(l0, l1), fmaxf(l2, l3));
  float e0 = __expf(l0 - mx), e1 = __expf(l1 - mx), e2 = __expf(l2 - mx), e3 = __expf(l3 - mx);
  float inv = 1.0f / (e0 + e1 + e2 + e3);
  int hh = l / WW, ww = l % WW;
  int lt = ww * HH + hh;
  size_t sstr = (size_t)LSEQ * DMODEL;
  float v = x[idx];
  v += e0 * inv * outs[(size_t)(0 * BATCH + b) * sstr + (size_t)l               * DMODEL + c];
  v += e1 * inv * outs[(size_t)(1 * BATCH + b) * sstr + (size_t)(LSEQ - 1 - l)  * DMODEL + c];
  v += e2 * inv * outs[(size_t)(2 * BATCH + b) * sstr + (size_t)lt              * DMODEL + c];
  v += e3 * inv * outs[(size_t)(3 * BATCH + b) * sstr + (size_t)(LSEQ - 1 - lt) * DMODEL + c];
  out[idx] = v;
}

extern "C" void kernel_launch(void* const* d_in, const int* in_sizes, int n_in,
                              void* d_out, int out_size, void* d_ws, size_t ws_size,
                              hipStream_t stream) {
  const float* x      = (const float*)d_in[0];
  const float* norm_w = (const float*)d_in[1];
  const float* logits = (const float*)d_in[2];
  const float* Win    = (const float*)d_in[3];
  const float* convW  = (const float*)d_in[4];
  const float* convb  = (const float*)d_in[5];
  const float* Wx     = (const float*)d_in[6];
  const float* Wdt    = (const float*)d_in[7];
  const float* bdt    = (const float*)d_in[8];
  const float* A_log  = (const float*)d_in[9];
  const float* Dp     = (const float*)d_in[10];
  const float* Wout   = (const float*)d_in[11];
  float* out = (float*)d_out;
  (void)in_sizes; (void)n_in; (void)out_size; (void)ws_size;

  char* ws = (char*)d_ws;
  size_t off = 0;
  auto alloc = [&](size_t bytes) -> char* {
    char* p = ws + off;
    off += (bytes + 255) & ~(size_t)255;
    return p;
  };
  _Float16* scans16 = (_Float16*)alloc((size_t)NSLICE * LSEQ * DMODEL * 2);
  _Float16* Win16   = (_Float16*)alloc((size_t)NSCAN * 2 * DINNER * DMODEL * 2);
  _Float16* Wx16    = (_Float16*)alloc((size_t)NSCAN * XDBLW * DINNER * 2);
  _Float16* Wout16  = (_Float16*)alloc((size_t)NSCAN * DMODEL * DINNER * 2);
  float*    xz      = (float*)  alloc((size_t)NSLICE * LSEQ * 2 * DINNER * 4);
  float*    u       = (float*)  alloc((size_t)NSLICE * LSEQ * DINNER * 4);
  _Float16* u16     = (_Float16*)alloc((size_t)NSLICE * LSEQ * DINNER * 2);
  float*    xdbl    = (float*)  alloc((size_t)NSLICE * LSEQ * XDBLW * 4);
  _Float16* y16     = (_Float16*)alloc((size_t)NSLICE * LSEQ * DINNER * 2);
  float*    outs    = (float*)  alloc((size_t)NSLICE * LSEQ * DMODEL * 4);

  // weight converts
  int nWin  = NSCAN * 2 * DINNER * DMODEL;
  int nWx   = NSCAN * XDBLW * DINNER;
  int nWout = NSCAN * DMODEL * DINNER;
  k_f2h<<<(nWin  + 255) / 256, 256, 0, stream>>>(Win,  Win16,  nWin);
  k_f2h<<<(nWx   + 255) / 256, 256, 0, stream>>>(Wx,   Wx16,   nWx);
  k_f2h<<<(nWout + 255) / 256, 256, 0, stream>>>(Wout, Wout16, nWout);

  // RMSNorm + scan construction
  k_rmsnorm_scan<<<BATCH * LSEQ, DMODEL, 0, stream>>>(x, norm_w, scans16);

  // in_proj: (L x 256) @ (1024 x 256)^T -> (L x 1024), 32x64 strip per wave
  dim3 g1(LSEQ / 32, (2 * DINNER) / 64, NSLICE);
  k_gemm_wmma<2, 4, 2 * DINNER, DMODEL><<<g1, 32, 0, stream>>>(scans16, Win16, xz, LSEQ);

  // depthwise conv + silu
  size_t tconv = (size_t)NSLICE * LSEQ * DINNER;
  k_conv_silu<<<(unsigned)((tconv + 255) / 256), 256, 0, stream>>>(xz, convW, convb, u, u16);

  // x_dbl: (L x 512) @ (48 x 512)^T -> (L x 48), 32x48 strip per wave
  dim3 g2(LSEQ / 32, 1, NSLICE);
  k_gemm_wmma<2, 3, XDBLW, DINNER><<<g2, 32, 0, stream>>>(u16, Wx16, xdbl, LSEQ);

  // sequential selective scan (fused dt projection, softplus, gating)
  k_scan<<<NSLICE, DINNER, 0, stream>>>(xdbl, u, xz, Wdt, bdt, A_log, Dp, y16);

  // out_proj: (L x 512) @ (256 x 512)^T -> (L x 256), 32x64 strip per wave
  dim3 g3(LSEQ / 32, DMODEL / 64, NSLICE);
  k_gemm_wmma<2, 4, DMODEL, DINNER><<<g3, 32, 0, stream>>>(y16, Wout16, outs, LSEQ);

  // combine + residual
  size_t tcomb = (size_t)BATCH * LSEQ * DMODEL;
  k_combine<<<(unsigned)((tcomb + 255) / 256), 256, 0, stream>>>(x, outs, logits, out);
}
